// MyNetRGCN_63522566307937
// MI455X (gfx1250) — compile-verified, matrix operand order checked by reading, not run on hardware
//
#include <hip/hip_runtime.h>
#include <hip/hip_bf16.h>

typedef __attribute__((ext_vector_type(16))) __bf16 v16bf;
typedef __attribute__((ext_vector_type(8)))  float  v8f;

#define B_   16
#define N_   4096
#define D_   256
#define E_   65536
#define KTOT 1280   // (R+1)*D_
#define VG_  30000
#define VS_  20000

static __device__ __forceinline__ __bf16 to_bf16(float f) { return (__bf16)f; }

// K-index inside a 32-deep bf16 WMMA operand for packed element j (0..15) held
// by a lane in half-group g (= lane>>4). Per CDNA5 ISA 7.12.2, 16-bit A 16x32:
// VGPR v<4: K = 2v+h (+8 for lanes 16-31); VGPR v>=4: K = 16+2(v-4)+h (+8).
static __device__ __forceinline__ int wmma_k(int j, int g) {
  int v = j >> 1, h = j & 1;
  int base = (v < 4) ? (2 * v) : (16 + 2 * (v - 4));
  return base + h + 8 * g;
}

// ---------------------------------------------------------------------------
// Stage 1: only edges with src==0 feed the output. Accumulate
//   y[b][r][:] = sum_{e: src=0, et=r} x[b, dst_e, :]   and deg[b][r],
// then Z[b] = [ x[b,0,:], y0/c0, y1/c1, y2/c2, y3/c3 ]  (length 1280).
// One block per graph; branch on src is wave-uniform; no atomics -> bitwise
// deterministic across graph replays.
// ---------------------------------------------------------------------------
__global__ void __launch_bounds__(256)
build_z_kernel(const float* __restrict__ x,
               const long long* __restrict__ edge_index,
               const long long* __restrict__ edge_type,
               float* __restrict__ z) {
  const int b = blockIdx.x;
  const int t = threadIdx.x;                      // feature component 0..255
  const long long* src = edge_index + (size_t)b * 2 * E_;
  const long long* dst = src + E_;
  const long long* rel = edge_type + (size_t)b * E_;
  const float* xb = x + (size_t)b * N_ * D_;
  float a0 = 0.f, a1 = 0.f, a2 = 0.f, a3 = 0.f;
  int   d0 = 0, d1 = 0, d2 = 0, d3 = 0;
  #pragma unroll 4
  for (int e = 0; e < E_; ++e) {
    if (src[e] == 0) {                            // uniform across the wave
      int j = (int)dst[e];
      int r = (int)rel[e];
      float v = xb[j * D_ + t];
      if      (r == 0) { a0 += v; ++d0; }
      else if (r == 1) { a1 += v; ++d1; }
      else if (r == 2) { a2 += v; ++d2; }
      else             { a3 += v; ++d3; }
    }
  }
  float* zb = z + b * KTOT;
  zb[t]           = xb[t];                        // self row, pairs with W_all[0]
  zb[D_ * 1 + t]  = a0 / (float)(d0 > 1 ? d0 : 1);
  zb[D_ * 2 + t]  = a1 / (float)(d1 > 1 ? d1 : 1);
  zb[D_ * 3 + t]  = a2 / (float)(d2 > 1 ? d2 : 1);
  zb[D_ * 4 + t]  = a3 / (float)(d3 > 1 ? d3 : 1);
}

// ---------------------------------------------------------------------------
// Stage 2: h0 = relu(Z @ Wbig), Z:[16,1280] bf16 (staged in LDS),
// Wbig = W_all flattened [1280,256]. Single workgroup, 16 waves, each wave
// owns one 16-column tile; 40 v_wmma_f32_16x16x32_bf16 k-steps.
// ---------------------------------------------------------------------------
__global__ void __launch_bounds__(512)
h0_gemm_kernel(const float* __restrict__ z, const float* __restrict__ W_all,
               __bf16* __restrict__ h0bf) {
  __shared__ __bf16 zs[B_ * KTOT];                // 40 KB LDS
  const int tid = threadIdx.x;
  for (int i = tid; i < B_ * KTOT; i += 512) zs[i] = to_bf16(z[i]);
  __syncthreads();

  const int wave = tid >> 5;                      // 16 waves (wave32)
  const int lane = tid & 31;
  const int g    = lane >> 4;
  const int l16  = lane & 15;
  const int ncol = wave * 16 + l16;               // output column 0..255
  int kk[16];
  #pragma unroll
  for (int j = 0; j < 16; ++j) kk[j] = wmma_k(j, g);

  v8f c = {};
  for (int k0 = 0; k0 < KTOT; k0 += 32) {
    v16bf a, bfr;
    #pragma unroll
    for (int j = 0; j < 16; ++j) {
      a[j]   = zs[l16 * KTOT + k0 + kk[j]];                 // A: M=l16
      bfr[j] = to_bf16(W_all[(k0 + kk[j]) * D_ + ncol]);    // B: N=l16-based col
    }
    c = __builtin_amdgcn_wmma_f32_16x16x32_bf16(false, a, false, bfr,
                                                (short)0, c, false, false);
  }
  #pragma unroll
  for (int i = 0; i < 8; ++i) {                   // C/D map: M = i + 8*g
    int m = i + 8 * g;                            // m = batch index
    float v = c[i] > 0.f ? c[i] : 0.f;            // ReLU
    h0bf[m * D_ + ncol] = to_bf16(v);
  }
}

// ---------------------------------------------------------------------------
// Stage 3: logits = h0 @ W + bias for one head. M=16(batch), K=256, N=V.
// 4 waves/WG, one 16-column tile per wave. The 32x16 fp32 k-slabs of W are
// staged into a per-wave LDS double buffer with coalesced
// global_load_async_to_lds_b128 (4 per slab per lane-set), pipelined via
// s_wait_asynccnt (async loads complete in order). Tail columns are handled
// by clamping the 4-column load group: WMMA columns are independent, so the
// clamped (garbage) columns are simply never written; EXEC stays all-ones
// around every WMMA as required.
// ---------------------------------------------------------------------------
#define HG_SLAB 32
__global__ void __launch_bounds__(128)
head_gemm_kernel(const __bf16* __restrict__ h0bf, const float* __restrict__ W,
                 const float* __restrict__ bias, float* __restrict__ logits,
                 const int V) {
  __shared__ float wtile[4][2][HG_SLAB * 16];     // 4 waves x dbl-buf x 2KB
  const int tid   = threadIdx.x;
  const int wave  = tid >> 5;
  const int lane  = tid & 31;
  const int g     = lane >> 4;
  const int l16   = lane & 15;
  const int nbase = (blockIdx.x * 4 + wave) * 16;
  const int n     = nbase + l16;

  // Cooperative slab copy coords: lane -> (row base, 4-column group).
  const int crow = lane >> 2;                     // 0..7 (x4 slabs of 8 rows)
  const int cgrp = lane & 3;                      // 0..3 -> cols cgrp*4..+3
  int col4 = nbase + cgrp * 4;
  if (col4 > V - 4) col4 = V - 4;                 // clamp, never masked
  const unsigned lds0 =
      (unsigned)(size_t)&wtile[wave][0][0];       // flat LDS addr[31:0] = offset
  const unsigned ldsStride = (unsigned)(HG_SLAB * 16 * sizeof(float));

  int kk[16];
  #pragma unroll
  for (int j = 0; j < 16; ++j) kk[j] = wmma_k(j, g);

  v16bf a_frag[8];
  #pragma unroll
  for (int s = 0; s < 8; ++s) {
    #pragma unroll
    for (int j = 0; j < 16; ++j)
      a_frag[s][j] = h0bf[l16 * D_ + s * 32 + kk[j]];       // A: M=l16=batch
  }

  auto stage = [&](int s) {                       // issue 4 async b128 loads
    const int p = s & 1;
    #pragma unroll
    for (int q = 0; q < 4; ++q) {
      const int row = q * 8 + crow;               // 0..31 within slab
      const unsigned long long gsrc =
          (unsigned long long)(size_t)(W + (size_t)(s * HG_SLAB + row) * V + col4);
      const unsigned ldst =
          lds0 + (unsigned)p * ldsStride +
          (unsigned)((row * 16 + cgrp * 4) * sizeof(float));
      asm volatile("global_load_async_to_lds_b128 %0, %1, off"
                   :: "v"(ldst), "v"(gsrc) : "memory");
    }
  };

  stage(0);
  v8f c = {};
  #pragma unroll
  for (int s = 0; s < 8; ++s) {
    if (s + 1 < 8) {
      stage(s + 1);                               // keep next slab in flight
      asm volatile("s_wait_asynccnt 0x4" ::: "memory");   // slab s complete
    } else {
      asm volatile("s_wait_asynccnt 0x0" ::: "memory");
    }
    const float* wl = &wtile[wave][s & 1][0];
    v16bf bfr;
    #pragma unroll
    for (int j = 0; j < 16; ++j)
      bfr[j] = to_bf16(wl[kk[j] * 16 + l16]);     // LDS -> B-frag (N=l16)
    c = __builtin_amdgcn_wmma_f32_16x16x32_bf16(false, a_frag[s], false, bfr,
                                                (short)0, c, false, false);
  }

  if (n < V) {
    float bv = bias[n];
    #pragma unroll
    for (int i = 0; i < 8; ++i) {
      int m = i + 8 * g;                          // batch row
      logits[m * V + n] = c[i] + bv;
    }
  }
}

// ---------------------------------------------------------------------------
// Stage 4: row-wise log_softmax, one block per (batch) row.
// ---------------------------------------------------------------------------
__global__ void __launch_bounds__(256)
logsoftmax_kernel(const float* __restrict__ logits, float* __restrict__ out,
                  const int V) {
  __shared__ float red[256];
  const int b = blockIdx.x;
  const int tid = threadIdx.x;
  const float* row = logits + (size_t)b * V;

  float m = -3.402823466e38f;
  for (int i = tid; i < V; i += 256) m = fmaxf(m, row[i]);
  red[tid] = m; __syncthreads();
  for (int s = 128; s > 0; s >>= 1) {
    if (tid < s) red[tid] = fmaxf(red[tid], red[tid + s]);
    __syncthreads();
  }
  const float mrow = red[0];
  __syncthreads();

  float acc = 0.f;
  for (int i = tid; i < V; i += 256) acc += expf(row[i] - mrow);
  red[tid] = acc; __syncthreads();
  for (int s = 128; s > 0; s >>= 1) {
    if (tid < s) red[tid] += red[tid + s];
    __syncthreads();
  }
  const float lse = mrow + logf(red[0]);
  __syncthreads();

  float* orow = out + (size_t)b * V;
  for (int i = tid; i < V; i += 256) orow[i] = row[i] - lse;
}

// ---------------------------------------------------------------------------
extern "C" void kernel_launch(void* const* d_in, const int* in_sizes, int n_in,
                              void* d_out, int out_size, void* d_ws, size_t ws_size,
                              hipStream_t stream) {
  const float*     x          = (const float*)d_in[0];
  const float*     W_all      = (const float*)d_in[1];
  const float*     Wg         = (const float*)d_in[2];
  const float*     bg         = (const float*)d_in[3];
  const float*     Ws         = (const float*)d_in[4];
  const float*     bs         = (const float*)d_in[5];
  const long long* edge_index = (const long long*)d_in[6];
  const long long* edge_type  = (const long long*)d_in[7];
  float* out = (float*)d_out;

  // Workspace carve (~3.3 MB total), 256B aligned slices. Every ws region we
  // read is fully written earlier in this same call.
  char* w = (char*)d_ws;
  auto carve = [&](size_t bytes) -> char* {
    char* p = w; w += (bytes + 255) & ~(size_t)255; return p;
  };
  float*  z       = (float*)carve(sizeof(float)  * B_ * KTOT);
  __bf16* h0bf    = (__bf16*)carve(sizeof(__bf16) * B_ * D_);
  float*  logitsg = (float*)carve(sizeof(float)  * B_ * VG_);
  float*  logitss = (float*)carve(sizeof(float)  * B_ * VS_);

  build_z_kernel<<<B_, 256, 0, stream>>>(x, edge_index, edge_type, z);
  h0_gemm_kernel<<<1, 512, 0, stream>>>(z, W_all, h0bf);
  head_gemm_kernel<<<(VG_ + 63) / 64, 128, 0, stream>>>(h0bf, Wg, bg, logitsg, VG_);
  head_gemm_kernel<<<(VS_ + 63) / 64, 128, 0, stream>>>(h0bf, Ws, bs, logitss, VS_);
  logsoftmax_kernel<<<B_, 256, 0, stream>>>(logitsg, out, VG_);
  logsoftmax_kernel<<<B_, 256, 0, stream>>>(logitss, out + (size_t)B_ * VG_, VS_);
}
